// AttentionBasedMultiModalFusion_66262755442886
// MI455X (gfx1250) — matrix operand order repaired and checked by reading, not compiled, not run on hardware
//
#include <hip/hip_runtime.h>
#include <stdint.h>

#define HH 300
#define DIMG 4096
#define DQ 300
#define VOCAB 8834
#define TIMG 50
#define TQ 30
#define NQ 128
#define STEPS 17

typedef __attribute__((ext_vector_type(16))) _Float16 v16h;
typedef __attribute__((ext_vector_type(8)))  float    v8f;

union H16pack { uint32_t u[8]; v16h v; };

__device__ __forceinline__ float sigm(float x) { return 1.f / (1.f + __expf(-x)); }

// ---------------------------------------------------------------------------
// WMMA GEMM with async-LDS B staging: C[M,N] = A[M,K] @ B[N,K]^T (+bias[n]).
//   A: f16, rows padded/zeroed to at least 128*gridDim.y, lda % 32 == 0.
//   B: f16 weights (K-major, torch [out,in] layout), rows padded to ceil64(N).
// Block = 256 threads = 8 waves. Each block owns a 64-wide N slab; wave w
// computes the (blockIdx.y*8+w)-th 16-row M tile over that slab.
// Per 32-K chunk the block DMAs the 64x32 B slab (4 KB) into LDS with
// GLOBAL_LOAD_ASYNC_TO_LDS_B128 (one 16B transfer per thread), double-buffered
// on ASYNCcnt so the DMA overlaps the WMMAs. B is thus fetched from L2 once
// per block instead of once per wave (8x traffic cut on the M=128 GEMMs).
// NOTE: the LDS address operand is derived from &sB[...] (ptrtoint) so the
// shared array escapes into the asm; otherwise LLVM treats sB as never-written
// and folds the B fragment loads to undef.
// ---------------------------------------------------------------------------
__global__ void __launch_bounds__(256) gemm_f16_wmma_lds(
    const _Float16* __restrict__ A, int lda,
    const _Float16* __restrict__ B, int ldb,
    float* __restrict__ C, int ldc,
    const float* __restrict__ bias,
    int M, int N, int K)
{
    __shared__ _Float16 sB[2][64 * 32];   // 2 x 4KB double buffer

    const int tid  = threadIdx.x;
    const int wave = tid >> 5;
    const int lane = tid & 31;
    const int hlf  = lane >> 4;   // 0/1
    const int l16  = lane & 15;
    const int mBase = (blockIdx.y * 8 + wave) * 16;
    const int nBase = blockIdx.x * 64;

    // async stager: thread i moves 16 bytes; slab row = i/4, 8-half segment = i%4
    // destination half-offset = i*8 (row*32 + seg*8 == i*8).
    const int srow = tid >> 2;
    const int sseg = tid & 3;
    const uint64_t gsrc0 =
        (uint64_t)(uintptr_t)(B + (size_t)(nBase + srow) * ldb + sseg * 8);

    v8f acc[4] = {};
    const uint32_t* Arow = (const uint32_t*)(A + (size_t)(mBase + l16) * lda);

    {   // prologue: stage chunk 0 into buffer 0
        uint32_t ldsAddr = (uint32_t)(uintptr_t)&sB[0][tid * 8];
        asm volatile("global_load_async_to_lds_b128 %0, %1, off"
                     :: "v"(ldsAddr), "v"(gsrc0) : "memory");
    }

    int buf = 0;
    for (int kc = 0; kc < K; kc += 32, buf ^= 1) {
        if (kc + 32 < K) {
            // prefetch next chunk into the other buffer, then wait for current
            uint32_t ldsAddr = (uint32_t)(uintptr_t)&sB[buf ^ 1][tid * 8];
            uint64_t g = gsrc0 + (uint64_t)(kc + 32) * 2;
            asm volatile("global_load_async_to_lds_b128 %0, %1, off"
                         :: "v"(ldsAddr), "v"(g) : "memory");
            asm volatile("s_wait_asynccnt 0x1" ::: "memory");
        } else {
            asm volatile("s_wait_asynccnt 0x0" ::: "memory");
        }
        __syncthreads();

        H16pack a;
#pragma unroll
        for (int i = 0; i < 8; ++i) {
            // A 16x32 f16 fragment: k = half*16 + i*2
            int k = kc + (hlf << 4) + (i << 1);
            a.u[i] = Arow[k >> 1];
        }
#pragma unroll
        for (int j = 0; j < 4; ++j) {
            // B 32x16 f16 fragment: row n = j*16+l16, k = half*16 + i*2
            const uint32_t* bs =
                (const uint32_t*)&sB[buf][(j * 16 + l16) * 32 + (hlf << 4)];
            H16pack b;
#pragma unroll
            for (int i = 0; i < 8; ++i) b.u[i] = bs[i];
            acc[j] = __builtin_amdgcn_wmma_f32_16x16x32_f16(
                false, a.v, false, b.v, (short)0, acc[j], false, false);
        }
        __syncthreads();   // protect buf before it is overwritten next round
    }

    const int mRow = mBase + hlf * 8;
#pragma unroll
    for (int r = 0; r < 8; ++r) {
        int m = mRow + r;
        if (m < M) {
#pragma unroll
            for (int j = 0; j < 4; ++j) {
                int n = nBase + j * 16 + l16;
                if (n < N) {
                    float v = acc[j][r];
                    if (bias) v += bias[n];
                    C[(size_t)m * ldc + n] = v;
                }
            }
        }
    }
}

// ---------------------------------------------------------------------------
// f32 -> f16 pack with optional column-concat of two sources, row/col zero pad.
// ---------------------------------------------------------------------------
__global__ void pack2_f16(_Float16* __restrict__ dst, int ldd, int Rp, int R,
                          const float* __restrict__ s1, int lds1, int co1, int C1,
                          const float* __restrict__ s2, int lds2, int co2, int C2)
{
    size_t total = (size_t)Rp * ldd;
    for (size_t idx = (size_t)blockIdx.x * blockDim.x + threadIdx.x; idx < total;
         idx += (size_t)gridDim.x * blockDim.x) {
        int r = (int)(idx / ldd), c = (int)(idx % ldd);
        float v = 0.f;
        if (r < R) {
            if (c < C1)            v = s1[(size_t)r * lds1 + co1 + c];
            else if (c < C1 + C2)  v = s2[(size_t)r * lds2 + co2 + (c - C1)];
        }
        dst[idx] = (_Float16)v;
    }
}

__global__ void zero_f32(float* p, size_t n) {
    for (size_t i = (size_t)blockIdx.x * blockDim.x + threadIdx.x; i < n;
         i += (size_t)gridDim.x * blockDim.x) p[i] = 0.f;
}
__global__ void zero_f16(_Float16* p, size_t n) {
    for (size_t i = (size_t)blockIdx.x * blockDim.x + threadIdx.x; i < n;
         i += (size_t)gridDim.x * blockDim.x) p[i] = (_Float16)0.f;
}
__global__ void copy2_f32(float* dst, const float* a, int na, const float* b, int nb) {
    int i = blockIdx.x * blockDim.x + threadIdx.x;
    if (i < na) dst[i] = a[i];
    else if (i < na + nb) dst[i] = b[i - na];
}

// v_i[c] = sum_n W_ai[n,300+c]*w_aih[n];  v_q[c] analog (the h-dependent score
// term is constant over positions, so softmax needs only these projections).
__global__ void vec_proj(float* vi, float* vq,
                         const float* __restrict__ Wai, const float* __restrict__ Waq,
                         const float* __restrict__ waih, const float* __restrict__ waqh)
{
    int c = blockIdx.x * blockDim.x + threadIdx.x;
    if (c < 600) {
        float s = 0.f;
        for (int n = 0; n < HH; ++n) s += Wai[(size_t)n * 900 + 300 + c] * waih[n];
        vi[c] = s;
    } else if (c < 1200) {
        int cc = c - 600;
        float s = 0.f;
        for (int n = 0; n < HH; ++n) s += Waq[(size_t)n * 900 + 300 + cc] * waqh[n];
        vq[cc] = s;
    }
}

// Ai[j] = W_ami[j,:]@ctx_i ; Fi[j] = W_fi[j,:]@ctx_i
__global__ void ctx_const(const float* __restrict__ Wami, const float* __restrict__ Wfi,
                          const float* __restrict__ ctx_i, float* Ai, float* Fi)
{
    int j = blockIdx.x * blockDim.x + threadIdx.x;
    if (j < 300) {
        float s = 0.f;
        for (int c = 0; c < 600; ++c) s += Wami[(size_t)j * 600 + c] * ctx_i[c];
        Ai[j] = s;
    } else if (j < 600) {
        int jj = j - 300;
        float s = 0.f;
        for (int c = 0; c < 600; ++c) s += Wfi[(size_t)jj * 600 + c] * ctx_i[c];
        Fi[jj] = s;
    }
}

// Video BiLSTM recurrence, batch 1, both directions; input projections (+bias)
// precomputed via WMMA GEMM into CinF/CinB [50,1200].
__global__ void __launch_bounds__(512) vid_recur(
    const float* __restrict__ CinF, const float* __restrict__ CinB,
    const float* __restrict__ WhhF, const float* __restrict__ WhhB,
    float* __restrict__ img_emb)
{
    __shared__ float hF[HH], hB[HH], cF[HH], cB[HH];
    __shared__ float gF[1200], gB[1200];
    int tid = threadIdx.x;
    for (int j = tid; j < HH; j += 512) { hF[j] = hB[j] = cF[j] = cB[j] = 0.f; }
    __syncthreads();
    for (int t = 0; t < TIMG; ++t) {
        for (int row = tid; row < 1200; row += 512) {
            float sf = CinF[t * 1200 + row], sb = CinB[t * 1200 + row];
            const float* wf = WhhF + (size_t)row * HH;
            const float* wb = WhhB + (size_t)row * HH;
            for (int k = 0; k < HH; ++k) { sf += wf[k] * hF[k]; sb += wb[k] * hB[k]; }
            gF[row] = sf; gB[row] = sb;
        }
        __syncthreads();
        if (tid < 300) {
            int j = tid;
            float i_ = sigm(gF[j]), f_ = sigm(gF[300 + j]);
            float g_ = tanhf(gF[600 + j]), o_ = sigm(gF[900 + j]);
            float c2 = f_ * cF[j] + i_ * g_; cF[j] = c2;
            float h2 = o_ * tanhf(c2); hF[j] = h2;
            img_emb[t * 600 + j] = h2;
        } else if (tid < 600) {
            int j = tid - 300;
            float i_ = sigm(gB[j]), f_ = sigm(gB[300 + j]);
            float g_ = tanhf(gB[600 + j]), o_ = sigm(gB[900 + j]);
            float c2 = f_ * cB[j] + i_ * g_; cB[j] = c2;
            float h2 = o_ * tanhf(c2); hB[j] = h2;
            img_emb[(TIMG - 1 - t) * 600 + 300 + j] = h2;
        }
        __syncthreads();
    }
}

// Build per-step question-LSTM inputs [x_t | h | pad] (f16) for both dirs.
__global__ void stage_q(_Float16* __restrict__ xF, _Float16* __restrict__ xB,
                        const float* __restrict__ qf,
                        const float* __restrict__ hF, const float* __restrict__ hB, int t)
{
    int idx = blockIdx.x * blockDim.x + threadIdx.x;
    if (idx >= NQ * 608) return;
    int q = idx / 608, c = idx % 608;
    float vF = 0.f, vB = 0.f;
    if (c < 300) {
        vF = qf[((size_t)q * TQ + t) * DQ + c];
        vB = qf[((size_t)q * TQ + (TQ - 1 - t)) * DQ + c];
    } else if (c < 600) {
        vF = hF[q * HH + (c - 300)];
        vB = hB[q * HH + (c - 300)];
    }
    xF[idx] = (_Float16)vF;
    xB[idx] = (_Float16)vB;
}

__global__ void q_update(const float* __restrict__ gFb, const float* __restrict__ gBb,
                         float* hF, float* hB, float* cF, float* cB,
                         float* __restrict__ q_emb, int t)
{
    int idx = blockIdx.x * blockDim.x + threadIdx.x;
    if (idx >= NQ * HH) return;
    int q = idx / HH, j = idx % HH;
    {
        const float* g = gFb + (size_t)q * 1200;
        float i_ = sigm(g[j]), f_ = sigm(g[300 + j]);
        float g_ = tanhf(g[600 + j]), o_ = sigm(g[900 + j]);
        float c2 = f_ * cF[idx] + i_ * g_; cF[idx] = c2;
        float h2 = o_ * tanhf(c2); hF[idx] = h2;
        q_emb[((size_t)q * TQ + t) * 600 + j] = h2;
    }
    {
        const float* g = gBb + (size_t)q * 1200;
        float i_ = sigm(g[j]), f_ = sigm(g[300 + j]);
        float g_ = tanhf(g[600 + j]), o_ = sigm(g[900 + j]);
        float c2 = f_ * cB[idx] + i_ * g_; cB[idx] = c2;
        float h2 = o_ * tanhf(c2); hB[idx] = h2;
        q_emb[((size_t)q * TQ + (TQ - 1 - t)) * 600 + 300 + j] = h2;
    }
}

// softmax(emb@v) pooled context; works for image (grid 1) and questions (grid 128)
__global__ void __launch_bounds__(256) attn_pool(
    const float* __restrict__ emb, const float* __restrict__ v,
    float* __restrict__ ctx, int T)
{
    __shared__ float s[64];
    int item = blockIdx.x, tid = threadIdx.x;
    const float* e = emb + (size_t)item * T * 600;
    if (tid < T) {
        float d = 0.f;
        for (int c = 0; c < 600; ++c) d += e[tid * 600 + c] * v[c];
        s[tid] = d;
    }
    __syncthreads();
    if (tid == 0) {
        float mx = s[0];
        for (int t = 1; t < T; ++t) mx = fmaxf(mx, s[t]);
        float sum = 0.f;
        for (int t = 0; t < T; ++t) { float ex = __expf(s[t] - mx); s[t] = ex; sum += ex; }
        float inv = 1.f / sum;
        for (int t = 0; t < T; ++t) s[t] *= inv;
    }
    __syncthreads();
    float* o = ctx + (size_t)item * 600;
    for (int c = tid; c < 600; c += 256) {
        float a = 0.f;
        for (int t = 0; t < T; ++t) a += s[t] * e[t * 600 + c];
        o[c] = a;
    }
}

// Modality-attention softmax(2) + fusion; writes fs (f16) into xcat[:,0:300].
// G1: [128,600]  cols 0-299 = W_am@h+b_am, 300-599 = W_f@h+b_f
// GQ: [128,600]  cols 0-299 = W_amq@ctx_q, 300-599 = W_fq@ctx_q
__global__ void __launch_bounds__(512) fuse_kernel(
    const float* __restrict__ G1, const float* __restrict__ Ai,
    const float* __restrict__ Fi, const float* __restrict__ GQ,
    const float* __restrict__ wamh, _Float16* __restrict__ xcat)
{
    __shared__ float r0[512], r1[512];
    int q = blockIdx.x, tid = threadIdx.x;
    float t0 = 0.f, t1 = 0.f;
    if (tid < 300) {
        float tmp = G1[(size_t)q * 600 + tid];
        t0 = wamh[tid] * tanhf(tmp + Ai[tid]);
        t1 = wamh[tid] * tanhf(tmp + GQ[(size_t)q * 600 + tid]);
    }
    r0[tid] = t0; r1[tid] = t1;
    __syncthreads();
    for (int s = 256; s > 0; s >>= 1) {
        if (tid < s) { r0[tid] += r0[tid + s]; r1[tid] += r1[tid + s]; }
        __syncthreads();
    }
    float e0 = r0[0], e1 = r1[0];
    float m = fmaxf(e0, e1);
    float x0 = __expf(e0 - m), x1 = __expf(e1 - m);
    float inv = 1.f / (x0 + x1);
    float mw0 = x0 * inv, mw1 = x1 * inv;
    if (tid < 300) {
        float fs = tanhf(G1[(size_t)q * 600 + 300 + tid] + mw0 * Fi[tid]
                         + mw1 * GQ[(size_t)q * 600 + 300 + tid]);
        xcat[(size_t)q * 928 + tid] = (_Float16)fs;
    }
}

// Decoder LSTM pointwise; writes h (f16) to h16 and into xcat[:,600:900].
__global__ void dec_update(const float* __restrict__ G2, float* __restrict__ c,
                           _Float16* __restrict__ h16, _Float16* __restrict__ xcat)
{
    int idx = blockIdx.x * blockDim.x + threadIdx.x;
    if (idx >= NQ * HH) return;
    int q = idx / HH, j = idx % HH;
    const float* g = G2 + (size_t)q * 1200;
    float i_ = sigm(g[j]), f_ = sigm(g[300 + j]);
    float g_ = tanhf(g[600 + j]), o_ = sigm(g[900 + j]);
    float c2 = f_ * c[idx] + i_ * g_; c[idx] = c2;
    float h2 = o_ * tanhf(c2);
    h16[(size_t)q * 320 + j] = (_Float16)h2;
    xcat[(size_t)q * 928 + 600 + j] = (_Float16)h2;
}

// Greedy argmax over logits row + glove gather (f16) into xcat[:,300:600].
__global__ void __launch_bounds__(256) argmax_glove(
    const float* __restrict__ out, int step,
    const float* __restrict__ glove, _Float16* __restrict__ xcat)
{
    __shared__ float bv[256];
    __shared__ int   bi[256];
    int q = blockIdx.x, tid = threadIdx.x;
    const float* row = out + (size_t)q * (STEPS * VOCAB) + (size_t)step * VOCAB;
    float best = -3.4e38f; int bidx = 0;
    for (int i = tid; i < VOCAB; i += 256) {
        float v = row[i];
        if (v > best) { best = v; bidx = i; }
    }
    bv[tid] = best; bi[tid] = bidx;
    __syncthreads();
    for (int s = 128; s > 0; s >>= 1) {
        if (tid < s) {
            if (bv[tid + s] > bv[tid] || (bv[tid + s] == bv[tid] && bi[tid + s] < bi[tid])) {
                bv[tid] = bv[tid + s]; bi[tid] = bi[tid + s];
            }
        }
        __syncthreads();
    }
    int idx = bi[0];
    for (int c = tid; c < DQ; c += 256)
        xcat[(size_t)q * 928 + 300 + c] = (_Float16)glove[(size_t)idx * DQ + c];
}

// ---------------------------------------------------------------------------
static inline void launch_gemm(const _Float16* A, int lda, const _Float16* B, int ldb,
                               float* C, int ldc, const float* bias,
                               int M, int N, int K, hipStream_t s)
{
    dim3 grid((N + 63) / 64, (M + 127) / 128);
    gemm_f16_wmma_lds<<<grid, 256, 0, s>>>(A, lda, B, ldb, C, ldc, bias, M, N, K);
}

static inline void launch_pack(_Float16* dst, int ldd, int Rp, int R,
                               const float* s1, int lds1, int co1, int C1,
                               const float* s2, int lds2, int co2, int C2, hipStream_t s)
{
    size_t total = (size_t)Rp * ldd;
    int blocks = (int)((total + 255) / 256);
    if (blocks > 8192) blocks = 8192;
    pack2_f16<<<blocks, 256, 0, s>>>(dst, ldd, Rp, R, s1, lds1, co1, C1, s2, lds2, co2, C2);
}

extern "C" void kernel_launch(void* const* d_in, const int* in_sizes, int n_in,
                              void* d_out, int out_size, void* d_ws, size_t ws_size,
                              hipStream_t stream)
{
    (void)in_sizes; (void)n_in; (void)out_size; (void)ws_size;

    const float* img_feats = (const float*)d_in[0];
    const float* q_feats   = (const float*)d_in[1];
    const float* glove     = (const float*)d_in[2];
    const float* vid_Wih_f = (const float*)d_in[3];
    const float* vid_Whh_f = (const float*)d_in[4];
    const float* vid_b_f   = (const float*)d_in[5];
    const float* vid_Wih_b = (const float*)d_in[6];
    const float* vid_Whh_b = (const float*)d_in[7];
    const float* vid_b_b   = (const float*)d_in[8];
    const float* que_Wih_f = (const float*)d_in[9];
    const float* que_Whh_f = (const float*)d_in[10];
    const float* que_b_f   = (const float*)d_in[11];
    const float* que_Wih_b = (const float*)d_in[12];
    const float* que_Whh_b = (const float*)d_in[13];
    const float* que_b_b   = (const float*)d_in[14];
    const float* W_ai      = (const float*)d_in[15];
    // d_in[16] = b_ai, d_in[18] = b_aq: shift-invariant under softmax -> unused
    const float* W_aq      = (const float*)d_in[17];
    const float* w_aih     = (const float*)d_in[19];
    const float* w_aqh     = (const float*)d_in[20];
    const float* W_am      = (const float*)d_in[21];
    const float* b_am      = (const float*)d_in[22];
    const float* W_ami     = (const float*)d_in[23];
    const float* W_amq     = (const float*)d_in[24];
    const float* w_amh     = (const float*)d_in[25];
    const float* W_fi      = (const float*)d_in[26];
    const float* W_fq      = (const float*)d_in[27];
    const float* W_f       = (const float*)d_in[28];
    const float* b_fuse    = (const float*)d_in[29];
    const float* dec_Wih   = (const float*)d_in[30];
    const float* dec_Whh   = (const float*)d_in[31];
    const float* dec_b     = (const float*)d_in[32];
    const float* W_out     = (const float*)d_in[33];
    const float* b_out     = (const float*)d_in[34];
    float* out = (float*)d_out;

    char* ws = (char*)d_ws;
    size_t off = 0;
    auto alloc = [&](size_t bytes) -> char* {
        char* p = ws + off;
        off += (bytes + 255) & ~(size_t)255;
        return p;
    };

    // f16 weight buffers (N padded to /64, K padded to /32)
    _Float16* vidWf16 = (_Float16*)alloc((size_t)1216 * 4096 * 2);
    _Float16* vidWb16 = (_Float16*)alloc((size_t)1216 * 4096 * 2);
    _Float16* queWf16 = (_Float16*)alloc((size_t)1216 * 608 * 2);  // [Wih|Whh]
    _Float16* queWb16 = (_Float16*)alloc((size_t)1216 * 608 * 2);
    _Float16* decW16  = (_Float16*)alloc((size_t)1216 * 928 * 2);  // [Wih(600)|Whh(300)]
    _Float16* W1cat   = (_Float16*)alloc((size_t)640 * 320 * 2);   // rows [W_am;W_f]
    _Float16* W2cat   = (_Float16*)alloc((size_t)640 * 608 * 2);   // rows [W_amq;W_fq]
    _Float16* Wout16  = (_Float16*)alloc((size_t)8896 * 320 * 2);
    _Float16* imgf16  = (_Float16*)alloc((size_t)128 * 4096 * 2);  // 128 rows for 8-wave blocks
    // f32 scratch
    float* biasG1  = (float*)alloc(600 * 4);                 // [b_am;b_f]
    float* vi      = (float*)alloc(600 * 4);
    float* vq      = (float*)alloc(600 * 4);
    float* CinF    = (float*)alloc((size_t)TIMG * 1200 * 4);
    float* CinB    = (float*)alloc((size_t)TIMG * 1200 * 4);
    float* img_emb = (float*)alloc((size_t)TIMG * 600 * 4);
    float* ctx_i   = (float*)alloc(600 * 4);
    float* Ai      = (float*)alloc(320 * 4);
    float* Fi      = (float*)alloc(320 * 4);
    _Float16* xcatF = (_Float16*)alloc((size_t)NQ * 608 * 2);
    _Float16* xcatB = (_Float16*)alloc((size_t)NQ * 608 * 2);
    float* gatesF  = (float*)alloc((size_t)NQ * 1200 * 4);
    float* gatesB  = (float*)alloc((size_t)NQ * 1200 * 4);
    // the following five must be consecutive (zeroed in one shot)
    float* hqF     = (float*)alloc((size_t)NQ * HH * 4);
    float* hqB     = (float*)alloc((size_t)NQ * HH * 4);
    float* cqF     = (float*)alloc((size_t)NQ * HH * 4);
    float* cqB     = (float*)alloc((size_t)NQ * HH * 4);
    float* cdec    = (float*)alloc((size_t)NQ * HH * 4);
    float* q_emb   = (float*)alloc((size_t)NQ * TQ * 600 * 4);
    float* ctx_q   = (float*)alloc((size_t)NQ * 600 * 4);
    _Float16* ctxq16 = (_Float16*)alloc((size_t)NQ * 608 * 2);
    float* GQ      = (float*)alloc((size_t)NQ * 600 * 4);
    // consecutive f16 state buffers (zeroed in one shot)
    _Float16* h16   = (_Float16*)alloc((size_t)NQ * 320 * 2);
    _Float16* xcatD = (_Float16*)alloc((size_t)NQ * 928 * 2);   // [fs|emb|h|pad]
    float* G1      = (float*)alloc((size_t)NQ * 600 * 4);
    float* G2      = (float*)alloc((size_t)NQ * 1200 * 4);

    // ---- weight packing (deterministic, every call) ----
    launch_pack(vidWf16, 4096, 1216, 1200, vid_Wih_f, 4096, 0, 4096, nullptr, 0, 0, 0, stream);
    launch_pack(vidWb16, 4096, 1216, 1200, vid_Wih_b, 4096, 0, 4096, nullptr, 0, 0, 0, stream);
    launch_pack(queWf16, 608, 1216, 1200, que_Wih_f, 300, 0, 300, que_Whh_f, 300, 0, 300, stream);
    launch_pack(queWb16, 608, 1216, 1200, que_Wih_b, 300, 0, 300, que_Whh_b, 300, 0, 300, stream);
    launch_pack(decW16, 928, 1216, 1200, dec_Wih, 600, 0, 600, dec_Whh, 300, 0, 300, stream);
    launch_pack(W1cat, 320, 300, 300, W_am, 300, 0, 300, nullptr, 0, 0, 0, stream);
    launch_pack(W1cat + (size_t)300 * 320, 320, 340, 300, W_f, 300, 0, 300, nullptr, 0, 0, 0, stream);
    launch_pack(W2cat, 608, 300, 300, W_amq, 600, 0, 600, nullptr, 0, 0, 0, stream);
    launch_pack(W2cat + (size_t)300 * 608, 608, 340, 300, W_fq, 600, 0, 600, nullptr, 0, 0, 0, stream);
    launch_pack(Wout16, 320, 8896, 8834, W_out, 300, 0, 300, nullptr, 0, 0, 0, stream);
    launch_pack(imgf16, 4096, 128, 50, img_feats, 4096, 0, 4096, nullptr, 0, 0, 0, stream);
    copy2_f32<<<3, 256, 0, stream>>>(biasG1, b_am, 300, b_fuse, 300);

    // ---- video path ----
    launch_gemm(imgf16, 4096, vidWf16, 4096, CinF, 1200, vid_b_f, TIMG, 1200, 4096, stream);
    launch_gemm(imgf16, 4096, vidWb16, 4096, CinB, 1200, vid_b_b, TIMG, 1200, 4096, stream);
    vid_recur<<<1, 512, 0, stream>>>(CinF, CinB, vid_Whh_f, vid_Whh_b, img_emb);
    vec_proj<<<5, 256, 0, stream>>>(vi, vq, W_ai, W_aq, w_aih, w_aqh);
    attn_pool<<<1, 256, 0, stream>>>(img_emb, vi, ctx_i, TIMG);
    ctx_const<<<3, 256, 0, stream>>>(W_ami, W_fi, ctx_i, Ai, Fi);

    // ---- zero state ----
    zero_f32<<<1024, 256, 0, stream>>>(hqF, (size_t)5 * NQ * HH);
    zero_f16<<<1024, 256, 0, stream>>>(h16, (size_t)NQ * 320 + (size_t)NQ * 928);

    // ---- question BiLSTM (batched over 128 questions) ----
    for (int t = 0; t < TQ; ++t) {
        stage_q<<<(NQ * 608 + 255) / 256, 256, 0, stream>>>(xcatF, xcatB, q_feats, hqF, hqB, t);
        launch_gemm(xcatF, 608, queWf16, 608, gatesF, 1200, que_b_f, NQ, 1200, 608, stream);
        launch_gemm(xcatB, 608, queWb16, 608, gatesB, 1200, que_b_b, NQ, 1200, 608, stream);
        q_update<<<(NQ * HH + 255) / 256, 256, 0, stream>>>(gatesF, gatesB, hqF, hqB, cqF, cqB, q_emb, t);
    }

    // ---- per-question attention context + fusion constants ----
    attn_pool<<<NQ, 256, 0, stream>>>(q_emb, vq, ctx_q, TQ);
    launch_pack(ctxq16, 608, NQ, NQ, ctx_q, 600, 0, 600, nullptr, 0, 0, 0, stream);
    launch_gemm(ctxq16, 608, W2cat, 608, GQ, 600, nullptr, NQ, 600, 608, stream);

    // ---- greedy decode, 17 sequential steps, batch 128 ----
    for (int st = 0; st < STEPS; ++st) {
        launch_gemm(h16, 320, W1cat, 320, G1, 600, biasG1, NQ, 600, 320, stream);
        fuse_kernel<<<NQ, 512, 0, stream>>>(G1, Ai, Fi, GQ, w_amh, xcatD);
        launch_gemm(xcatD, 928, decW16, 928, G2, 1200, dec_b, NQ, 1200, 928, stream);
        dec_update<<<(NQ * HH + 255) / 256, 256, 0, stream>>>(G2, cdec, h16, xcatD);
        launch_gemm(h16, 320, Wout16, 320, out + (size_t)st * VOCAB, STEPS * VOCAB,
                    b_out, NQ, VOCAB, 320, stream);
        argmax_glove<<<NQ, 256, 0, stream>>>(out, st, glove, xcatD);
    }
}
